// Generator_87875030876561
// MI455X (gfx1250) — matrix operand order repaired and checked by reading, not compiled, not run on hardware
//
#include <hip/hip_runtime.h>
#include <math.h>

// ---------------------------------------------------------------------------
// MI455X (gfx1250) implementation notes:
//  - All dense layers -> v_wmma_f32_16x16x32_f16 (wave32, 16x16 tiles, f32 acc)
//  - Activations kept in L2 (every tensor <= 16MB, L2 = 192MB)
//  - Bias/activation/residual fused into WMMA epilogues
//  - Edge aggregation (PointGNN scatter-mean, LINKX SparseLinear) via
//    global_atomic_add_f32
//  - Tiny 128->{1,3,4} heads / BN stats / max-pool as VALU reductions
//  - GEMM streams use addrspace(1) loads (global_load_b128, LOADcnt-only
//    waits) with pointer-increment addressing (immediate-offset folding)
// ---------------------------------------------------------------------------

typedef __attribute__((ext_vector_type(16))) _Float16     v16h;
typedef __attribute__((ext_vector_type(8)))  float        v8f;
typedef __attribute__((ext_vector_type(4)))  unsigned int v4u;

// explicit global-address-space 128-bit load (forces global_load_b128,
// avoids flat_load_b128 + DScnt coupling); plain ext-vector type so the
// addrspace(1) lvalue copies without a class operator=
typedef __attribute__((address_space(1))) const v4u gq4_t;
#define GLOAD4(p) (*(gq4_t*)(p))

#define NN 16384
#define NE 262144

// ---------------------------------------------------------------------------
// Weight packing: B-tile layout so each lane reads its 16 halves contiguously.
// packed[((nt*KT + kt)*32 + lane)*16 + i] = W[kt*32 + (lane>=16)*16 + i,
//                                             nt*16 + (lane&15)]
// ---------------------------------------------------------------------------
__global__ void pack_w_kernel(const float* __restrict__ w, _Float16* __restrict__ dst,
                              int din, int dout, int dinp) {
  int idx = blockIdx.x * blockDim.x + threadIdx.x;
  int total = dinp * dout;
  if (idx >= total) return;
  int i    = idx & 15;
  int lane = (idx >> 4) & 31;
  int tile = idx >> 9;
  int KT   = dinp >> 5;
  int kt   = tile % KT;
  int nt   = tile / KT;
  int K    = kt * 32 + ((lane >> 4) & 1) * 16 + i;
  int col  = nt * 16 + (lane & 15);
  float v = (K < din) ? w[(size_t)K * dout + col] : 0.0f;
  dst[idx] = (_Float16)v;
}

// ---------------------------------------------------------------------------
// WMMA GEMM: out[n, j] = act(sum_k X[n,k] * W[k,j] + bias[j]) (+res1)(+res2)
// Each wave computes a 16(M) x 64(N) strip: 4 accumulator tiles, A reused 4x.
// act: 0=none 1=relu 2=leaky(alpha)
// ---------------------------------------------------------------------------
__global__ __launch_bounds__(256)
void gemm_wmma_kernel(const _Float16* __restrict__ X, int lda,
                      const _Float16* __restrict__ PW,
                      const float* __restrict__ bias,
                      float* __restrict__ outf, _Float16* __restrict__ outh,
                      const float* __restrict__ res1, const float* __restrict__ res2,
                      int nrows, int dout, int act, float alpha) {
  const int lane = threadIdx.x & 31;
  const int wid  = threadIdx.x >> 5;
  const int NT4  = dout >> 6;                       // groups of 4 N-tiles
  const int task = blockIdx.x * 8 + wid;
  const int mtiles = nrows >> 4;
  if (task >= mtiles * NT4) return;
  const int mt = task / NT4;
  const int ng = task % NT4;
  const int m0 = mt << 4;
  const int n0 = ng << 6;
  const int KT = lda >> 5;
  const int row  = m0 + (lane & 15);
  const int hsel = (lane >> 4) & 1;

  union HV { v16h h; v4u q[2]; };

  v8f acc[4];
#pragma unroll
  for (int t = 0; t < 4; ++t) {
    float bv = bias ? bias[n0 + t * 16 + (lane & 15)] : 0.0f;
#pragma unroll
    for (int r = 0; r < 8; ++r) acc[t][r] = bv;
  }

  // Streaming pointers: A advances 32 halves (64B) per K-step; each B tile
  // advances 512 halves (1024B) per K-step.
  const _Float16* ap = X + (size_t)row * lda + hsel * 16;
  const _Float16* bp[4];
#pragma unroll
  for (int t = 0; t < 4; ++t)
    bp[t] = PW + ((size_t)(((n0 >> 4) + t) * KT) * 32 + lane) * 16;

  for (int kt = 0; kt < KT; ++kt) {
    HV a;
    a.q[0] = GLOAD4(ap);
    a.q[1] = GLOAD4(ap + 8);
    if (kt + 1 < KT) __builtin_prefetch((const void*)(ap + 32), 0, 1);
    ap += 32;
#pragma unroll
    for (int t = 0; t < 4; ++t) {
      HV b;
      b.q[0] = GLOAD4(bp[t]);
      b.q[1] = GLOAD4(bp[t] + 8);
      bp[t] += 512;
      acc[t] = __builtin_amdgcn_wmma_f32_16x16x32_f16(
          false, a.h, false, b.h, (short)0, acc[t], false, false);
    }
  }

  const int mrow  = m0 + 8 * hsel;
  const int ncol0 = n0 + (lane & 15);
#pragma unroll
  for (int t = 0; t < 4; ++t) {
#pragma unroll
    for (int r = 0; r < 8; ++r) {
      int m = mrow + r;
      int n = ncol0 + t * 16;
      float v = acc[t][r];
      if (act == 1)      v = fmaxf(v, 0.0f);
      else if (act == 2) v = (v > 0.0f) ? v : alpha * v;
      size_t idx = (size_t)m * dout + n;
      if (res1) v += res1[idx];
      if (res2) v += res2[idx];
      if (outf) outf[idx] = v;
      if (outh) outh[idx] = (_Float16)v;
    }
  }
}

// ---------------------------------------------------------------------------
// Fourier feature encoding: x = [cos(2*pi*pos@B^T), sin(...)]  -> [N,128]
// ---------------------------------------------------------------------------
__global__ void encode_kernel(const float* __restrict__ pos, const float* __restrict__ B,
                              float* __restrict__ xf, _Float16* __restrict__ xh) {
  int idx = blockIdx.x * blockDim.x + threadIdx.x;
  if (idx >= NN * 64) return;
  int n = idx >> 6, j = idx & 63;
  float vp = 6.2831853071795864f *
             (pos[n * 3 + 0] * B[j * 3 + 0] +
              pos[n * 3 + 1] * B[j * 3 + 1] +
              pos[n * 3 + 2] * B[j * 3 + 2]);
  float s, c;
  sincosf(vp, &s, &c);
  xf[(size_t)n * 128 + j]      = c;
  xf[(size_t)n * 128 + 64 + j] = s;
  xh[(size_t)n * 128 + j]      = (_Float16)c;
  xh[(size_t)n * 128 + 64 + j] = (_Float16)s;
}

// small 128 -> dout linear + tanh (used for mlp_h second layer and tail2)
__global__ void small_lin_tanh_kernel(const float* __restrict__ in, const float* __restrict__ w,
                                      const float* __restrict__ b, float* __restrict__ out,
                                      int dout) {
  int idx = blockIdx.x * blockDim.x + threadIdx.x;
  if (idx >= NN * dout) return;
  int n = idx / dout, c = idx % dout;
  float s = b[c];
  const float* ir = in + (size_t)n * 128;
  for (int k = 0; k < 128; ++k) s += ir[k] * w[k * dout + c];
  out[idx] = tanhf(s);
}

__global__ void zero_kernel(float* __restrict__ p, int count) {
  int idx = blockIdx.x * blockDim.x + threadIdx.x;
  if (idx < count) p[idx] = 0.0f;
}

// PointGNN message scatter: agg[dst] += [pos_j-pos_i+delta_i, x_j], cnt[dst]++
__global__ void pg_scatter_kernel(const int* __restrict__ ei, const float* __restrict__ pos,
                                  const float* __restrict__ delta, const float* __restrict__ xf,
                                  float* __restrict__ agg, float* __restrict__ cnt) {
  int e = blockIdx.x * blockDim.x + threadIdx.x;
  if (e >= NE) return;
  int s = ei[e], d = ei[NE + e];
  atomicAdd(&cnt[d], 1.0f);
  float* ap = agg + (size_t)d * 160;
  for (int c = 0; c < 3; ++c)
    atomicAdd(&ap[c], pos[s * 3 + c] - pos[d * 3 + c] + delta[d * 3 + c]);
  const float* xr = xf + (size_t)s * 128;
  for (int k = 0; k < 128; ++k) atomicAdd(&ap[3 + k], xr[k]);
}

__global__ void mean_half_kernel(const float* __restrict__ agg, const float* __restrict__ cnt,
                                 _Float16* __restrict__ aggh) {
  int idx = blockIdx.x * blockDim.x + threadIdx.x;
  if (idx >= NN * 160) return;
  int n = idx / 160;
  float ct = fmaxf(cnt[n], 1.0f);
  aggh[idx] = (_Float16)(agg[idx] / ct);
}

// column max over [N,128] -> h128  (one block per column)
__global__ void colmax_kernel(const float* __restrict__ xf, float* __restrict__ h128) {
  __shared__ float sm[256];
  int col = blockIdx.x;
  float m = -INFINITY;
  for (int r = threadIdx.x; r < NN; r += 256) m = fmaxf(m, xf[(size_t)r * 128 + col]);
  sm[threadIdx.x] = m;
  __syncthreads();
  for (int s = 128; s > 0; s >>= 1) {
    if (threadIdx.x < s) sm[threadIdx.x] = fmaxf(sm[threadIdx.x], sm[threadIdx.x + s]);
    __syncthreads();
  }
  if (threadIdx.x == 0) h128[col] = sm[0];
}

// global feature: hg = leaky_relu(h @ glob_w + glob_b, 0.2)
__global__ void glob_kernel(const float* __restrict__ h, const float* __restrict__ w,
                            const float* __restrict__ b, float* __restrict__ hg) {
  int j = threadIdx.x;
  if (j >= 128) return;
  float s = b[j];
  for (int k = 0; k < 128; ++k) s += h[k] * w[k * 128 + j];
  hg[j] = (s > 0.0f) ? s : 0.2f * s;
}

__global__ void concat256_kernel(const float* __restrict__ xf, const float* __restrict__ hg,
                                 float* __restrict__ x256, _Float16* __restrict__ x256h) {
  int idx = blockIdx.x * blockDim.x + threadIdx.x;
  if (idx >= NN * 256) return;
  int n = idx >> 8, j = idx & 255;
  float v = (j < 128) ? xf[(size_t)n * 128 + j] : hg[j - 128];
  x256[idx]  = v;
  x256h[idx] = (_Float16)v;
}

// LINKX SparseLinear: out[dst] += A_w[src]  (64 features per thread)
__global__ void lx_scatter_kernel(const int* __restrict__ ei, const float* __restrict__ Aw,
                                  float* __restrict__ out) {
  int t = blockIdx.x * blockDim.x + threadIdx.x;
  if (t >= NE * 4) return;
  int e = t >> 2, ch = (t & 3) * 64;
  int s = ei[e], d = ei[NE + e];
  const float* ar = Aw + (size_t)s * 256 + ch;
  float* op = out + (size_t)d * 256 + ch;
  for (int k = 0; k < 64; ++k) atomicAdd(&op[k], ar[k]);
}

__global__ void bias_half_kernel(float* __restrict__ out, const float* __restrict__ Ab,
                                 _Float16* __restrict__ outh) {
  int idx = blockIdx.x * blockDim.x + threadIdx.x;
  if (idx >= NN * 256) return;
  float v = out[idx] + Ab[idx & 255];
  out[idx]  = v;
  outh[idx] = (_Float16)v;
}

__global__ void relu_half_kernel(const float* __restrict__ in, _Float16* __restrict__ out,
                                 int count) {
  int idx = blockIdx.x * blockDim.x + threadIdx.x;
  if (idx < count) out[idx] = (_Float16)fmaxf(in[idx], 0.0f);
}

// batch-norm stats over nodes (population var) -- one block per column
__global__ void bnstats_kernel(const float* __restrict__ h, float* __restrict__ mu,
                               float* __restrict__ var) {
  __shared__ float s1[256], s2[256];
  int col = blockIdx.x;
  float a = 0.0f, b = 0.0f;
  for (int r = threadIdx.x; r < NN; r += 256) {
    float v = h[(size_t)r * 256 + col];
    a += v;
    b += v * v;
  }
  s1[threadIdx.x] = a;
  s2[threadIdx.x] = b;
  __syncthreads();
  for (int s = 128; s > 0; s >>= 1) {
    if (threadIdx.x < s) { s1[threadIdx.x] += s1[threadIdx.x + s]; s2[threadIdx.x] += s2[threadIdx.x + s]; }
    __syncthreads();
  }
  if (threadIdx.x == 0) {
    float m = s1[0] / (float)NN;
    mu[col]  = m;
    var[col] = s2[0] / (float)NN - m * m;
  }
}

__global__ void bnapply_kernel(const float* __restrict__ h, const float* __restrict__ mu,
                               const float* __restrict__ var, const float* __restrict__ g,
                               const float* __restrict__ b, _Float16* __restrict__ hh) {
  int idx = blockIdx.x * blockDim.x + threadIdx.x;
  if (idx >= NN * 256) return;
  int c = idx & 255;
  float v = (h[idx] - mu[c]) * rsqrtf(var[c] + 1e-5f) * g[c] + b[c];
  hh[idx] = (_Float16)v;
}

__global__ void cat512_kernel(const float* __restrict__ g, const _Float16* __restrict__ x256h,
                              _Float16* __restrict__ cat) {
  int idx = blockIdx.x * blockDim.x + threadIdx.x;
  if (idx >= NN * 512) return;
  int n = idx >> 9, j = idx & 511;
  cat[idx] = (j < 256) ? (_Float16)g[(size_t)n * 256 + j] : x256h[(size_t)n * 256 + (j - 256)];
}

// heads: scaling(softplus,3), rotation(norm,4), opacity(sigmoid,1), shs(3), xyz(3)
__global__ void heads_kernel(const float* __restrict__ f, const float* __restrict__ pc,
                             const float* wsc, const float* bsc,
                             const float* wrot, const float* brot,
                             const float* wop, const float* bop,
                             const float* wshs, const float* bshs,
                             const float* wxyz, const float* bxyz,
                             float* __restrict__ out) {
  int n = blockIdx.x * blockDim.x + threadIdx.x;
  if (n >= NN) return;
  float sc[3] = {bsc[0], bsc[1], bsc[2]};
  float rt[4] = {brot[0], brot[1], brot[2], brot[3]};
  float op    = bop[0];
  float sh[3] = {bshs[0], bshs[1], bshs[2]};
  float xy[3] = {bxyz[0], bxyz[1], bxyz[2]};
  const float* fr = f + (size_t)n * 128;
  for (int k = 0; k < 128; ++k) {
    float fv = fr[k];
    for (int c = 0; c < 3; ++c) {
      sc[c] += fv * wsc[k * 3 + c];
      sh[c] += fv * wshs[k * 3 + c];
      xy[c] += fv * wxyz[k * 3 + c];
    }
    for (int c = 0; c < 4; ++c) rt[c] += fv * wrot[k * 4 + c];
    op += fv * wop[k];
  }
  float nr = sqrtf(rt[0]*rt[0] + rt[1]*rt[1] + rt[2]*rt[2] + rt[3]*rt[3]);
  nr = fmaxf(nr, 1e-12f);
  for (int c = 0; c < 3; ++c) {
    float s = 1.0f / (1.0f + expf(-xy[c]));
    out[(size_t)n * 3 + c] = (s - 0.5f) * (1.2f / 32.0f) + pc[n * 3 + c];   // xyz
    out[(size_t)8 * NN + n * 3 + c]  = log1pf(expf(sc[c]));                 // scaling
    out[(size_t)11 * NN + n * 3 + c] = sh[c];                               // shs
  }
  out[(size_t)3 * NN + n] = 1.0f / (1.0f + expf(-op));                      // opacity
  for (int c = 0; c < 4; ++c) out[(size_t)4 * NN + n * 4 + c] = rt[c] / nr; // rotation
}

// ---------------------------------------------------------------------------
extern "C" void kernel_launch(void* const* d_in, const int* in_sizes, int n_in,
                              void* d_out, int out_size, void* d_ws, size_t ws_size,
                              hipStream_t stream) {
  (void)in_sizes; (void)n_in; (void)out_size; (void)ws_size;
  auto F = [&](int i) { return (const float*)d_in[i]; };
  const float* pos = F(0);
  const int*   ei  = (const int*)d_in[1];
  // jax pytree flatten order (alphabetical keys, 'b' < 'w'):
  const float* pB      = F(3);
  const float* dec1b = F(4),  *dec1w = F(5);
  const float* dec2b = F(6),  *dec2w = F(7);
  const float* globb = F(8),  *globw = F(9);
  // lx1: 10..23, lx2: 24..37
  const float *lxAb[2], *lxAw[2], *lxbnb[2], *lxbng[2], *lxc1b[2], *lxc1w[2],
              *lxc2b[2], *lxc2w[2], *lxf1b[2], *lxf1w[2], *lxf2b[2], *lxf2w[2],
              *lxndb[2], *lxndw[2];
  for (int i = 0; i < 2; ++i) {
    int o = 10 + 14 * i;
    lxAb[i] = F(o); lxAw[i] = F(o+1); lxbnb[i] = F(o+2); lxbng[i] = F(o+3);
    lxc1b[i] = F(o+4); lxc1w[i] = F(o+5); lxc2b[i] = F(o+6); lxc2w[i] = F(o+7);
    lxf1b[i] = F(o+8); lxf1w[i] = F(o+9); lxf2b[i] = F(o+10); lxf2w[i] = F(o+11);
    lxndb[i] = F(o+12); lxndw[i] = F(o+13);
  }
  const float* opab = F(38), *opaw = F(39);
  const float *pgg1b[2], *pgg1w[2], *pgg2b[2], *pgg2w[2], *pgh1b[2], *pgh1w[2],
              *pgh2b[2], *pgh2w[2];
  for (int i = 0; i < 2; ++i) {
    int o = 40 + 8 * i;
    pgg1b[i] = F(o); pgg1w[i] = F(o+1); pgg2b[i] = F(o+2); pgg2w[i] = F(o+3);
    pgh1b[i] = F(o+4); pgh1w[i] = F(o+5); pgh2b[i] = F(o+6); pgh2w[i] = F(o+7);
  }
  const float* rotb = F(56), *rotw = F(57);
  const float* scab = F(58), *scaw = F(59);
  const float* shsb = F(60), *shsw = F(61);
  const float* t1b  = F(62), *t1w  = F(63);
  const float* t2b  = F(64), *t2w  = F(65);
  const float* xyzb = F(66), *xyzw = F(67);

  char* wsb = (char*)d_ws;
  const size_t MB_ = (size_t)1 << 20;
  // stage-1 arena
  const size_t P_XF = 0, P_XH = 8*MB_, P_T1 = 12*MB_, P_T1H = 20*MB_;
  const size_t P_DELTA = 24*MB_, P_CNT = 24*MB_ + 262144, P_AGG = 25*MB_, P_AGGH = 35*MB_;
  // LINKX arena (reuses stage-1 space)
  const size_t L_OUT = 0, L_OUTH = 16*MB_, L_OUT2 = 24*MB_, L_XN = 40*MB_, L_XNH = 56*MB_;
  const size_t L_OUT3 = 64*MB_, L_H = 80*MB_, L_HH = 96*MB_, L_G = 104*MB_, L_GH = 120*MB_;
  // decode arena (reuses LINKX dead space)
  const size_t D_CAT = 0, D_D1H = 16*MB_, D_F = 24*MB_;
  // persistent
  const size_t P_X256 = 128*MB_, P_X256H = 144*MB_, P_PC = 152*MB_;
  const size_t P_SM = P_PC + 262144, P_PW = 153*MB_;

  auto F32 = [&](size_t o) { return (float*)(wsb + o); };
  auto F16 = [&](size_t o) { return (_Float16*)(wsb + o); };
  float* h128 = F32(P_SM);
  float* hg   = F32(P_SM + 512);
  float* mu   = F32(P_SM + 1024);
  float* var  = F32(P_SM + 2048);

  // --- pack all GEMM weights to f16 WMMA tile layout ---
  _Float16* pwbase = F16(P_PW);
  size_t pwoff = 0;
  auto pack = [&](const float* w, int din, int dout, int dinp) -> const _Float16* {
    _Float16* dst = pwbase + pwoff;
    pwoff += (size_t)dinp * dout;
    int total = dinp * dout;
    pack_w_kernel<<<(total + 255) / 256, 256, 0, stream>>>(w, dst, din, dout, dinp);
    return dst;
  };
  const _Float16 *pw_h1[2], *pw_g1[2], *pw_g2[2];
  for (int i = 0; i < 2; ++i) {
    pw_h1[i] = pack(pgh1w[i], 128, 128, 128);
    pw_g1[i] = pack(pgg1w[i], 131, 128, 160);   // K padded 131 -> 160
    pw_g2[i] = pack(pgg2w[i], 128, 128, 128);
  }
  const _Float16* pw_tail1 = pack(t1w, 256, 128, 256);
  const _Float16 *pw_node[2], *pw_cat1[2], *pw_cat2[2], *pw_f1[2], *pw_f2[2];
  for (int i = 0; i < 2; ++i) {
    pw_node[i] = pack(lxndw[i], 256, 256, 256);
    pw_cat1[i] = pack(lxc1w[i], 256, 256, 256);
    pw_cat2[i] = pack(lxc2w[i], 256, 256, 256);
    pw_f1[i]   = pack(lxf1w[i], 256, 256, 256);
    pw_f2[i]   = pack(lxf2w[i], 256, 256, 256);
  }
  const _Float16* pw_dec1 = pack(dec1w, 512, 256, 512);
  const _Float16* pw_dec2 = pack(dec2w, 256, 128, 256);

  auto gemm = [&](const _Float16* X, int lda, const _Float16* PW, const float* bias,
                  float* outf, _Float16* outh, const float* r1, const float* r2,
                  int dout, int act, float alpha) {
    int blocks = ((NN / 16) * (dout / 64)) / 8;
    gemm_wmma_kernel<<<blocks, 256, 0, stream>>>(X, lda, PW, bias, outf, outh, r1, r2,
                                                 NN, dout, act, alpha);
  };

  // --- Fourier feature encoding ---
  encode_kernel<<<(NN * 64) / 256, 256, 0, stream>>>(pos, pB, F32(P_XF), F16(P_XH));

  // --- two PointGNNConv blocks ---
  for (int i = 0; i < 2; ++i) {
    gemm(F16(P_XH), 128, pw_h1[i], pgh1b[i], F32(P_T1), nullptr, nullptr, nullptr, 128, 1, 0.f);
    small_lin_tanh_kernel<<<(NN * 3 + 255) / 256, 256, 0, stream>>>(
        F32(P_T1), pgh2w[i], pgh2b[i], F32(P_DELTA), 3);
    zero_kernel<<<(NN * 160 + 255) / 256, 256, 0, stream>>>(F32(P_AGG), NN * 160);
    zero_kernel<<<(NN + 255) / 256, 256, 0, stream>>>(F32(P_CNT), NN);
    pg_scatter_kernel<<<NE / 256, 256, 0, stream>>>(ei, pos, F32(P_DELTA), F32(P_XF),
                                                    F32(P_AGG), F32(P_CNT));
    mean_half_kernel<<<(NN * 160) / 256, 256, 0, stream>>>(F32(P_AGG), F32(P_CNT), F16(P_AGGH));
    gemm(F16(P_AGGH), 160, pw_g1[i], pgg1b[i], nullptr, F16(P_T1H), nullptr, nullptr, 128, 1, 0.f);
    // out = relu(g2(...)); x = x + out   (act-then-residual epilogue, in-place)
    gemm(F16(P_T1H), 128, pw_g2[i], pgg2b[i], F32(P_XF), F16(P_XH), F32(P_XF), nullptr, 128, 1, 0.f);
  }

  // --- global max pool + glob linear + concat -> x256 ---
  colmax_kernel<<<128, 256, 0, stream>>>(F32(P_XF), h128);
  glob_kernel<<<1, 128, 0, stream>>>(h128, globw, globb, hg);
  concat256_kernel<<<(NN * 256) / 256, 256, 0, stream>>>(F32(P_XF), hg,
                                                         F32(P_X256), F16(P_X256H));

  // --- tail: pc = tanh(tail2(leaky(tail1(x256), 0.2))) ---
  gemm(F16(P_X256H), 256, pw_tail1, t1b, F32(P_XF), nullptr, nullptr, nullptr, 128, 2, 0.2f);
  small_lin_tanh_kernel<<<(NN * 3 + 255) / 256, 256, 0, stream>>>(
      F32(P_XF), t2w, t2b, F32(P_PC), 3);

  // --- two LINKX blocks ---
  const _Float16* xin_h = F16(P_X256H);
  for (int i = 0; i < 2; ++i) {
    zero_kernel<<<(NN * 256 + 255) / 256, 256, 0, stream>>>(F32(L_OUT), NN * 256);
    lx_scatter_kernel<<<(NE * 4) / 256, 256, 0, stream>>>(ei, lxAw[i], F32(L_OUT));
    bias_half_kernel<<<(NN * 256) / 256, 256, 0, stream>>>(F32(L_OUT), lxAb[i], F16(L_OUTH));
    // out2 = out + cat1(out)
    gemm(F16(L_OUTH), 256, pw_cat1[i], lxc1b[i], F32(L_OUT2), nullptr, F32(L_OUT), nullptr, 256, 0, 0.f);
    // xn = node(x)
    gemm(xin_h, 256, pw_node[i], lxndb[i], F32(L_XN), F16(L_XNH), nullptr, nullptr, 256, 0, 0.f);
    // out3 = out2 + xn + cat2(xn)
    gemm(F16(L_XNH), 256, pw_cat2[i], lxc2b[i], F32(L_OUT3), nullptr, F32(L_OUT2), F32(L_XN), 256, 0, 0.f);
    relu_half_kernel<<<(NN * 256) / 256, 256, 0, stream>>>(F32(L_OUT3), F16(L_OUTH), NN * 256);
    // h = relu(f1(relu(out3)))
    gemm(F16(L_OUTH), 256, pw_f1[i], lxf1b[i], F32(L_H), nullptr, nullptr, nullptr, 256, 1, 0.f);
    bnstats_kernel<<<256, 256, 0, stream>>>(F32(L_H), mu, var);
    bnapply_kernel<<<(NN * 256) / 256, 256, 0, stream>>>(F32(L_H), mu, var, lxbng[i], lxbnb[i], F16(L_HH));
    // g = f2(h_bn)
    gemm(F16(L_HH), 256, pw_f2[i], lxf2b[i], F32(L_G), F16(L_GH), nullptr, nullptr, 256, 0, 0.f);
    xin_h = F16(L_GH);
  }

  // --- decode: cat([g, x256]) -> dec1 -> dec2 -> heads ---
  cat512_kernel<<<(NN * 512) / 256, 256, 0, stream>>>(F32(L_G), F16(P_X256H), F16(D_CAT));
  gemm(F16(D_CAT), 512, pw_dec1, dec1b, nullptr, F16(D_D1H), nullptr, nullptr, 256, 2, 0.01f);
  gemm(F16(D_D1H), 256, pw_dec2, dec2b, F32(D_F), nullptr, nullptr, nullptr, 128, 2, 0.01f);
  heads_kernel<<<(NN + 255) / 256, 256, 0, stream>>>(
      F32(D_F), F32(P_PC), scaw, scab, rotw, rotb, opaw, opab, shsw, shsb, xyzw, xyzb,
      (float*)d_out);
}
